// BahdanauAttention_15753940041975
// MI455X (gfx1250) — compile-verified, matrix operand order checked by reading, not run on hardware
//
#include <hip/hip_runtime.h>

#define NLAYERS 2
#define NB 32
#define NS 2048
#define NH 1024
#define TS 64          // S-rows per tile in the scores kernel
#define LDSK 1032      // padded bf16 elements per LDS row (2064B stride -> conflict-free)

typedef __attribute__((ext_vector_type(16))) __bf16 v16bf;
typedef __attribute__((ext_vector_type(8)))  float  v8f;
typedef __attribute__((ext_vector_type(4)))  int    v4i;

union Frag { v16bf bf; v4i q[2]; };
struct Frags { Frag a0, a1, b0, b1; };

__device__ __forceinline__ unsigned short f2bf(float f) {
  unsigned int u = __float_as_uint(f);
  u += 0x7fffu + ((u >> 16) & 1u);          // round-to-nearest-even
  return (unsigned short)(u >> 16);
}
__device__ __forceinline__ unsigned int pk_bf16(float a, float b) {
  return (unsigned int)f2bf(a) | ((unsigned int)f2bf(b) << 16);
}
__device__ __forceinline__ float fast_tanh(float x) {
#if __has_builtin(__builtin_amdgcn_tanhf)
  return __builtin_amdgcn_tanhf(x);         // v_tanh_f32 (gfx1250 trans op)
#else
  float cx = fminf(fmaxf(x, -15.f), 15.f);
  float e  = __expf(2.f * cx);
  return (e - 1.f) * __builtin_amdgcn_rcpf(e + 1.f);  // avoid full IEEE divide
#endif
}

// ---------------------------------------------------------------- Wk -> bf16
__global__ __launch_bounds__(256)
void cvt_wk_kernel(const float* __restrict__ Wk, unsigned int* __restrict__ dst) {
  const int i = blockIdx.x * 256 + threadIdx.x;      // over float4s: NH*NH/4
  float4 v = ((const float4*)Wk)[i];
  uint2 p;
  p.x = pk_bf16(v.x, v.y);
  p.y = pk_bf16(v.z, v.w);
  ((uint2*)dst)[i] = p;
}

// ------------------------------------------- qb[b,h] = h[-1]@WqT + bq + bk
__global__ __launch_bounds__(256)
void qb_kernel(const float* __restrict__ hidden, const float* __restrict__ Wq,
               const float* __restrict__ bq, const float* __restrict__ bk,
               float* __restrict__ qb) {
  __shared__ float sh[NH];
  const int tid = threadIdx.x;
  const int idx = blockIdx.x * 256 + tid;            // 0..NB*NH-1
  const int b = idx >> 10;                           // 256 | 1024 -> uniform b per block
  const int h = idx & (NH - 1);
  const float* hv = hidden + (size_t)(NLAYERS - 1) * NB * NH + (size_t)b * NH;
  for (int i = tid; i < NH; i += 256) sh[i] = hv[i];
  __syncthreads();
  const float* wr = Wq + (size_t)h * NH;
  float acc = 0.f;
  #pragma unroll 4
  for (int k = 0; k < NH; k += 4) {
    float4 w = *(const float4*)(wr + k);
    acc = fmaf(sh[k + 0], w.x, acc);
    acc = fmaf(sh[k + 1], w.y, acc);
    acc = fmaf(sh[k + 2], w.z, acc);
    acc = fmaf(sh[k + 3], w.w, acc);
  }
  qb[idx] = acc + bq[h] + bk[h];
}

// ---------- scores[b,s] = wv . tanh(qb[b,:] + enc[b,s,:]@WkT) + bv  (WMMA)
// Wave mapping: 8 waves = 2 (M, 32 rows via 2 A-frags) x 4 (N, 32 cols via 2 B-frags).
// Per K-step: 4 ds_load_b128 + 4 global_load_b128 -> 4 WMMAs (2x2 fragment tile),
// double-buffered so loads for step k+1 are in flight during the WMMAs of step k.
__global__ __launch_bounds__(256)
void scores_kernel(const float* __restrict__ enc,
                   const unsigned short* __restrict__ WkBf,  // [NH][NH] bf16, row n over k
                   const float* __restrict__ qb,             // [NB][NH]
                   const float* __restrict__ wv,             // [NH]
                   const float* __restrict__ bv,             // [1]
                   float* __restrict__ scores) {             // [NB][NS]
  extern __shared__ unsigned short lds[];                    // [TS][LDSK] bf16 enc tile
  __shared__ float sScores[TS];
  const int b   = blockIdx.y;
  const int s0  = blockIdx.x * TS;
  const int tid = threadIdx.x;

  if (tid < TS) sScores[tid] = 0.f;

  // cooperative load + f32->bf16 convert of the enc tile (TS x NH)
  const float* src = enc + ((size_t)b * NS + s0) * NH;
  for (int i = tid; i < TS * (NH / 4); i += 256) {
    const int row = i >> 8;                    // NH/4 = 256
    const int c4  = i & 255;
    float4 v = ((const float4*)(src + (size_t)row * NH))[c4];
    uint2 p;
    p.x = pk_bf16(v.x, v.y);
    p.y = pk_bf16(v.z, v.w);
    ((uint2*)(lds + (size_t)row * LDSK))[c4] = p;
  }
  __syncthreads();

  const int wave = tid >> 5;                   // wave32
  const int lane = tid & 31;
  const int mi   = wave >> 2;                  // 0..1 : rows mi*32 .. mi*32+31
  const int ni   = wave & 3;                   // 0..3 : 32 cols per nit iteration
  const int l15  = lane & 15;
  const int koff = (lane >> 4) * 8;            // K sub-offset per lane half

  const unsigned short* aPtr = lds + (size_t)(mi * 32 + l15) * LDSK + koff;

  float acc0[8] = {0.f, 0.f, 0.f, 0.f, 0.f, 0.f, 0.f, 0.f};  // rows mi*32 + 0..15
  float acc1[8] = {0.f, 0.f, 0.f, 0.f, 0.f, 0.f, 0.f, 0.f};  // rows mi*32 + 16..31

  #define LOAD_FRAGS(F, BP, K)                                            \
    do {                                                                  \
      (F).a0.q[0] = *(const v4i*)(aPtr + (K));                            \
      (F).a0.q[1] = *(const v4i*)(aPtr + (K) + 16);                       \
      (F).a1.q[0] = *(const v4i*)(aPtr + (K) + 16 * LDSK);                \
      (F).a1.q[1] = *(const v4i*)(aPtr + (K) + 16 * LDSK + 16);           \
      (F).b0.q[0] = *(const v4i*)((BP) + (K));                            \
      (F).b0.q[1] = *(const v4i*)((BP) + (K) + 16);                       \
      (F).b1.q[0] = *(const v4i*)((BP) + (K) + 16 * NH);                  \
      (F).b1.q[1] = *(const v4i*)((BP) + (K) + 16 * NH + 16);             \
    } while (0)

  #define WMMA4(F)                                                        \
    do {                                                                  \
      c00 = __builtin_amdgcn_wmma_f32_16x16x32_bf16(false, (F).a0.bf, false, (F).b0.bf, (short)0, c00, false, false); \
      c01 = __builtin_amdgcn_wmma_f32_16x16x32_bf16(false, (F).a0.bf, false, (F).b1.bf, (short)0, c01, false, false); \
      c10 = __builtin_amdgcn_wmma_f32_16x16x32_bf16(false, (F).a1.bf, false, (F).b0.bf, (short)0, c10, false, false); \
      c11 = __builtin_amdgcn_wmma_f32_16x16x32_bf16(false, (F).a1.bf, false, (F).b1.bf, (short)0, c11, false, false); \
    } while (0)

  for (int nit = 0; nit < NH / 128; ++nit) {           // 8 iterations
    const int n0 = nit * 128 + ni * 32;
    const int nA = n0 + l15;                           // b0 column for this lane
    const int nB = nA + 16;                            // b1 column for this lane
    const unsigned short* bPtr = WkBf + (size_t)nA * NH + koff;

    v8f c00 = {0.f, 0.f, 0.f, 0.f, 0.f, 0.f, 0.f, 0.f};
    v8f c01 = {0.f, 0.f, 0.f, 0.f, 0.f, 0.f, 0.f, 0.f};
    v8f c10 = {0.f, 0.f, 0.f, 0.f, 0.f, 0.f, 0.f, 0.f};
    v8f c11 = {0.f, 0.f, 0.f, 0.f, 0.f, 0.f, 0.f, 0.f};

    Frags f0, f1;
    LOAD_FRAGS(f0, bPtr, 0);
    for (int kk = 0; kk < 15; ++kk) {                  // K = 1024 in steps of 64
      LOAD_FRAGS(f1, bPtr, kk * 64 + 32);
      WMMA4(f0);
      LOAD_FRAGS(f0, bPtr, kk * 64 + 64);
      WMMA4(f1);
    }
    LOAD_FRAGS(f1, bPtr, NH - 32);
    WMMA4(f0);
    WMMA4(f1);

    // epilogue: e = tanh(C + qb[n]); acc += wv[n] * e
    const float qA = qb[(size_t)b * NH + nA];
    const float qB = qb[(size_t)b * NH + nB];
    const float wA = wv[nA];
    const float wB = wv[nB];
    #pragma unroll
    for (int v = 0; v < 8; ++v) {
      acc0[v] += wA * fast_tanh(c00[v] + qA) + wB * fast_tanh(c01[v] + qB);
      acc1[v] += wA * fast_tanh(c10[v] + qA) + wB * fast_tanh(c11[v] + qB);
    }
  }
  #undef LOAD_FRAGS
  #undef WMMA4

  // reduce over the 16 N-lanes inside each lane-half, then accumulate per-row
  #pragma unroll
  for (int v = 0; v < 8; ++v) {
    float x0 = acc0[v], x1 = acc1[v];
    for (int msk = 1; msk <= 8; msk <<= 1) {
      x0 += __shfl_xor(x0, msk, 32);
      x1 += __shfl_xor(x1, msk, 32);
    }
    acc0[v] = x0;
    acc1[v] = x1;
  }
  if (l15 == 0) {
    const int half = lane >> 4;
    #pragma unroll
    for (int v = 0; v < 8; ++v) {
      atomicAdd(&sScores[mi * 32 + v + 8 * half], acc0[v]);        // ds_add_f32
      atomicAdd(&sScores[mi * 32 + 16 + v + 8 * half], acc1[v]);
    }
  }
  __syncthreads();

  if (tid < TS)
    scores[(size_t)b * NS + s0 + tid] = sScores[tid] + bv[0];
}

// ---------------------------------------------------- softmax over S, in place
__global__ __launch_bounds__(256)
void softmax_kernel(float* __restrict__ scores) {
  __shared__ float red[8];
  const int b = blockIdx.x, tid = threadIdx.x;
  float* row = scores + (size_t)b * NS;

  float m = -1e30f;
  #pragma unroll
  for (int j = 0; j < NS / 256; ++j) m = fmaxf(m, row[tid + j * 256]);
  for (int o = 16; o; o >>= 1) m = fmaxf(m, __shfl_xor(m, o, 32));
  if ((tid & 31) == 0) red[tid >> 5] = m;
  __syncthreads();
  float bm = red[0];
  #pragma unroll
  for (int w = 1; w < 8; ++w) bm = fmaxf(bm, red[w]);
  __syncthreads();

  float vals[NS / 256];
  float s = 0.f;
  #pragma unroll
  for (int j = 0; j < NS / 256; ++j) {
    vals[j] = __expf(row[tid + j * 256] - bm);
    s += vals[j];
  }
  for (int o = 16; o; o >>= 1) s += __shfl_xor(s, o, 32);
  if ((tid & 31) == 0) red[tid >> 5] = s;
  __syncthreads();
  float bs = 0.f;
  #pragma unroll
  for (int w = 0; w < 8; ++w) bs += red[w];
  const float inv = 1.0f / bs;
  #pragma unroll
  for (int j = 0; j < NS / 256; ++j) row[tid + j * 256] = vals[j] * inv;
}

// ------------------------------------------- out[b,h] = sum_s attn[b,s]*enc[b,s,h]
__global__ __launch_bounds__(256)
void out_kernel(const float* __restrict__ enc, const float* __restrict__ attn,
                float* __restrict__ out) {
  __shared__ float sa[NS];
  const int tid = threadIdx.x;
  const int idx = blockIdx.x * 256 + tid;   // 0..NB*NH-1
  const int b = idx >> 10;                  // uniform per block
  const int h = idx & (NH - 1);
  const float* arow = attn + (size_t)b * NS;
  for (int i = tid; i < NS; i += 256) sa[i] = arow[i];
  __syncthreads();
  const float* e = enc + (size_t)b * NS * NH + h;
  float a0 = 0.f, a1 = 0.f, a2 = 0.f, a3 = 0.f;
  for (int s = 0; s < NS; s += 4) {
    a0 = fmaf(sa[s + 0], e[(size_t)(s + 0) << 10], a0);
    a1 = fmaf(sa[s + 1], e[(size_t)(s + 1) << 10], a1);
    a2 = fmaf(sa[s + 2], e[(size_t)(s + 2) << 10], a2);
    a3 = fmaf(sa[s + 3], e[(size_t)(s + 3) << 10], a3);
  }
  out[idx] = (a0 + a1) + (a2 + a3);
}

// --------------------------------------------------------------------- launch
extern "C" void kernel_launch(void* const* d_in, const int* in_sizes, int n_in,
                              void* d_out, int out_size, void* d_ws, size_t ws_size,
                              hipStream_t stream) {
  const float* hidden = (const float*)d_in[0];
  const float* enc    = (const float*)d_in[1];
  const float* Wq     = (const float*)d_in[2];
  const float* bq     = (const float*)d_in[3];
  const float* Wk     = (const float*)d_in[4];
  const float* bk     = (const float*)d_in[5];
  const float* wv     = (const float*)d_in[6];
  const float* bv     = (const float*)d_in[7];
  float* out = (float*)d_out;

  char* ws = (char*)d_ws;
  unsigned short* WkBf = (unsigned short*)ws;                          // 2 MB
  float* qb     = (float*)(ws + (size_t)2 * 1024 * 1024);              // 128 KB
  float* scores = (float*)(ws + (size_t)2 * 1024 * 1024 + 131072);     // 256 KB (reused as attn)

  cvt_wk_kernel<<<(NH * NH / 4) / 256, 256, 0, stream>>>(Wk, (unsigned int*)WkBf);
  qb_kernel<<<(NB * NH) / 256, 256, 0, stream>>>(hidden, Wq, bq, bk, qb);
  scores_kernel<<<dim3(NS / TS, NB), 256, (size_t)TS * LDSK * sizeof(unsigned short), stream>>>(
      enc, WkBf, qb, wv, bv, scores);
  softmax_kernel<<<NB, 256, 0, stream>>>(scores);
  out_kernel<<<(NB * NH) / 256, 256, 0, stream>>>(enc, scores, out);
}
